// GA_13580686590562
// MI455X (gfx1250) — matrix operand order repaired and checked by reading, not compiled, not run on hardware
//
#include <hip/hip_runtime.h>
#include <hip/hip_bf16.h>
#include <math.h>

typedef __attribute__((ext_vector_type(16))) _Float16 v16h;
typedef __attribute__((ext_vector_type(8)))  _Float16 v8h;
typedef __attribute__((ext_vector_type(8)))  float    v8f;

static constexpr int Bn = 8;
static constexpr int Cn = 512;
static constexpr int Tn = 2048;
static constexpr float TEMP = 0.07f;
static constexpr float EPS  = 1e-12f;

// ---------------------------------------------------------------------------
// WMMA GEMM: D[M,N] = A[M,K] * B^T[N,K]   (both operands K-major rows)
//   A: lda elements per M-row, contiguous in K
//   B: ldb elements per N-row, contiguous in K
//   D: ldd elements per M-row, contiguous in N
// Each wave computes a 64x16 block: 4 M-tiles sharing one B fragment.
// K must be a multiple of 64: the K-loop is unrolled x2 with two register
// fragment buffers so next-chunk loads overlap current-chunk WMMAs.
// ---------------------------------------------------------------------------
union Frag16 {
    v16h v;
    v8h  h[2];
};

#define LOAD_A(frag, pa, koff)                                        \
    (frag).h[0] = *(const v8h*)((pa) + (koff) + aoff0;                \
    (frag).h[1] = *(const v8h*)((pa) + (koff) + aoff1)

template <bool STORE_HALF>
__global__ __launch_bounds__(128) void wmma_gemm_f16(
    const _Float16* __restrict__ A, const _Float16* __restrict__ B,
    void* __restrict__ D, int M, int N, int K, int lda, int ldb, int ldd)
{
    const int lane  = threadIdx.x & 31;
    const int wave  = threadIdx.x >> 5;
    const int tilesM = M >> 6;                     // 64-row M blocks
    const int tile  = blockIdx.x * 4 + wave;
    const int m0 = (tile % tilesM) << 6;
    const int n0 = (tile / tilesM) << 4;
    const int half = lane >> 4;                    // 16-lane half of the wave
    const int row  = lane & 15;

    // Per-lane row pointers, K-major.
    const _Float16* pa0 = A + (size_t)(m0 + row) * lda;
    const _Float16* pa1 = pa0 + (size_t)16 * lda;
    const _Float16* pa2 = pa0 + (size_t)32 * lda;
    const _Float16* pa3 = pa0 + (size_t)48 * lda;
    const _Float16* pb  = B + (size_t)(n0 + row) * ldb;

    // Fragment sub-offsets (in halves):
    //   A 16x32 layout: half h -> K = h*8+[0..7] and 16+h*8+[0..7]
    //   B 32x16 layout: half h -> K = h*8+[0..7] and 16+h*8+[0..7]
    const int aoff0 = half * 8;
    const int aoff1 = 16 + half * 8;
    const int boff  = half * 8;

    v8f acc0 = {}, acc1 = {}, acc2 = {}, acc3 = {};

    Frag16 a0, a1, a2, a3, fb;      // buffer X (even chunks)
    Frag16 c0, c1, c2, c3, fd;      // buffer Y (odd chunks)

    // Preload chunk k=0 into buffer X.
    a0.h[0] = *(const v8h*)(pa0 + aoff0); a0.h[1] = *(const v8h*)(pa0 + aoff1);
    a1.h[0] = *(const v8h*)(pa1 + aoff0); a1.h[1] = *(const v8h*)(pa1 + aoff1);
    a2.h[0] = *(const v8h*)(pa2 + aoff0); a2.h[1] = *(const v8h*)(pa2 + aoff1);
    a3.h[0] = *(const v8h*)(pa3 + aoff0); a3.h[1] = *(const v8h*)(pa3 + aoff1);
    fb.h[0] = *(const v8h*)(pb + boff);   fb.h[1] = *(const v8h*)(pb + boff + 16);

    for (int k0 = 0; k0 < K; k0 += 64) {
        // Issue loads for chunk k0+32 into buffer Y (always valid: K % 64 == 0).
        {
            const int k1 = k0 + 32;
            c0.h[0] = *(const v8h*)(pa0 + k1 + aoff0); c0.h[1] = *(const v8h*)(pa0 + k1 + aoff1);
            c1.h[0] = *(const v8h*)(pa1 + k1 + aoff0); c1.h[1] = *(const v8h*)(pa1 + k1 + aoff1);
            c2.h[0] = *(const v8h*)(pa2 + k1 + aoff0); c2.h[1] = *(const v8h*)(pa2 + k1 + aoff1);
            c3.h[0] = *(const v8h*)(pa3 + k1 + aoff0); c3.h[1] = *(const v8h*)(pa3 + k1 + aoff1);
            fd.h[0] = *(const v8h*)(pb + k1 + boff);   fd.h[1] = *(const v8h*)(pb + k1 + boff + 16);
        }
        // 4 independent WMMAs on buffer X (overlaps with Y loads in flight).
        acc0 = __builtin_amdgcn_wmma_f32_16x16x32_f16(false, a0.v, false, fb.v, (short)0, acc0, false, false);
        acc1 = __builtin_amdgcn_wmma_f32_16x16x32_f16(false, a1.v, false, fb.v, (short)0, acc1, false, false);
        acc2 = __builtin_amdgcn_wmma_f32_16x16x32_f16(false, a2.v, false, fb.v, (short)0, acc2, false, false);
        acc3 = __builtin_amdgcn_wmma_f32_16x16x32_f16(false, a3.v, false, fb.v, (short)0, acc3, false, false);

        // Issue loads for chunk k0+64 into buffer X (skip on last iteration).
        if (k0 + 64 < K) {
            const int k2 = k0 + 64;
            a0.h[0] = *(const v8h*)(pa0 + k2 + aoff0); a0.h[1] = *(const v8h*)(pa0 + k2 + aoff1);
            a1.h[0] = *(const v8h*)(pa1 + k2 + aoff0); a1.h[1] = *(const v8h*)(pa1 + k2 + aoff1);
            a2.h[0] = *(const v8h*)(pa2 + k2 + aoff0); a2.h[1] = *(const v8h*)(pa2 + k2 + aoff1);
            a3.h[0] = *(const v8h*)(pa3 + k2 + aoff0); a3.h[1] = *(const v8h*)(pa3 + k2 + aoff1);
            fb.h[0] = *(const v8h*)(pb + k2 + boff);   fb.h[1] = *(const v8h*)(pb + k2 + boff + 16);
        }
        // 4 independent WMMAs on buffer Y.
        acc0 = __builtin_amdgcn_wmma_f32_16x16x32_f16(false, c0.v, false, fd.v, (short)0, acc0, false, false);
        acc1 = __builtin_amdgcn_wmma_f32_16x16x32_f16(false, c1.v, false, fd.v, (short)0, acc1, false, false);
        acc2 = __builtin_amdgcn_wmma_f32_16x16x32_f16(false, c2.v, false, fd.v, (short)0, acc2, false, false);
        acc3 = __builtin_amdgcn_wmma_f32_16x16x32_f16(false, c3.v, false, fd.v, (short)0, acc3, false, false);
    }

    // C/D f32 layout: VGPR r -> M = (sub-tile base) + r + 8*half, N = n0 + row
    if (STORE_HALF) {
        _Float16* Dh = (_Float16*)D;
#pragma unroll
        for (int r = 0; r < 8; ++r) {
            const size_t mrow = (size_t)(m0 + r + 8 * half);
            Dh[(mrow)      * ldd + n0 + row] = (_Float16)acc0[r];
            Dh[(mrow + 16) * ldd + n0 + row] = (_Float16)acc1[r];
            Dh[(mrow + 32) * ldd + n0 + row] = (_Float16)acc2[r];
            Dh[(mrow + 48) * ldd + n0 + row] = (_Float16)acc3[r];
        }
    } else {
        float* Df = (float*)D;
#pragma unroll
        for (int r = 0; r < 8; ++r) {
            const size_t mrow = (size_t)(m0 + r + 8 * half);
            Df[(mrow)      * ldd + n0 + row] = acc0[r];
            Df[(mrow + 16) * ldd + n0 + row] = acc1[r];
            Df[(mrow + 32) * ldd + n0 + row] = acc2[r];
            Df[(mrow + 48) * ldd + n0 + row] = acc3[r];
        }
    }
}

// ---------------------------------------------------------------------------
// Elementwise / reduction helpers
// ---------------------------------------------------------------------------

// flat f32 -> f16 convert (weights)
__global__ void cvt_f32_to_f16(const float* __restrict__ in,
                               _Float16* __restrict__ out, int n)
{
    int i = blockIdx.x * 256 + threadIdx.x;
    if (i < n) out[i] = (_Float16)in[i];
}

// x[b] in [C,T] f32 -> xT [T,C] f16 (coalesced on the write side)
__global__ void transpose_to_f16(const float* __restrict__ x,
                                 _Float16* __restrict__ xT)
{
    int i = blockIdx.x * 256 + threadIdx.x;   // i over T*C
    int t = i / Cn;
    int c = i - t * Cn;
    xT[i] = (_Float16)x[(size_t)c * Tn + t];
}

// depthwise conv k=3 pad=1 over T for all 3C channels (f32)
__global__ void dwconv3(const float* __restrict__ in,
                        const float* __restrict__ wdw,   // [3C,1,3] flat
                        float* __restrict__ out)
{
    int i = blockIdx.x * 256 + threadIdx.x;   // over 3C*T
    int t  = i & (Tn - 1);
    int ch = i >> 11;                          // / Tn
    const float* w = wdw + ch * 3;
    const float* p = in + (size_t)ch * Tn;
    float l = (t > 0)      ? p[t - 1] : 0.f;
    float m = p[t];
    float r = (t < Tn - 1) ? p[t + 1] : 0.f;
    out[i] = w[0] * l + w[1] * m + w[2] * r;
}

// per-token (t) channel L2-norm of q,k; pack qT,kT as [T,C] f16 and v as [C,T] f16
__global__ void norm_pack(const float* __restrict__ conv,  // [3C,T] f32
                          _Float16* __restrict__ qnT,      // [T,C]
                          _Float16* __restrict__ knT,      // [T,C]
                          _Float16* __restrict__ vh)       // [C,T]
{
    const int t   = blockIdx.x;
    const int tid = threadIdx.x;
    __shared__ float sq[256], sk[256];
    float aq = 0.f, ak = 0.f;
    for (int c = tid; c < Cn; c += 256) {
        float q = conv[(size_t)c * Tn + t];
        float k = conv[(size_t)(Cn + c) * Tn + t];
        aq += q * q;
        ak += k * k;
    }
    sq[tid] = aq; sk[tid] = ak;
    __syncthreads();
    for (int s = 128; s > 0; s >>= 1) {
        if (tid < s) { sq[tid] += sq[tid + s]; sk[tid] += sk[tid + s]; }
        __syncthreads();
    }
    const float rq = 1.f / fmaxf(sqrtf(sq[0]), EPS);
    const float rk = 1.f / fmaxf(sqrtf(sk[0]), EPS);
    for (int c = tid; c < Cn; c += 256) {
        qnT[(size_t)t * Cn + c] = (_Float16)(conv[(size_t)c * Tn + t] * rq);
        knT[(size_t)t * Cn + c] = (_Float16)(conv[(size_t)(Cn + c) * Tn + t] * rk);
        vh[(size_t)c * Tn + t]  = (_Float16)conv[(size_t)(2 * Cn + c) * Tn + t];
    }
}

// row softmax of S/TEMP -> f16 P
__global__ void softmax_row(const float* __restrict__ S,
                            _Float16* __restrict__ P)
{
    const int t   = blockIdx.x;
    const int tid = threadIdx.x;
    const float* row = S + (size_t)t * Tn;
    __shared__ float red[256];

    float m = -3.4e38f;
    for (int s = tid; s < Tn; s += 256) m = fmaxf(m, row[s]);
    red[tid] = m; __syncthreads();
    for (int s = 128; s > 0; s >>= 1) {
        if (tid < s) red[tid] = fmaxf(red[tid], red[tid + s]);
        __syncthreads();
    }
    m = red[0]; __syncthreads();

    const float invT = 1.f / TEMP;
    float sum = 0.f;
    for (int s = tid; s < Tn; s += 256) sum += expf((row[s] - m) * invT);
    red[tid] = sum; __syncthreads();
    for (int s = 128; s > 0; s >>= 1) {
        if (tid < s) red[tid] += red[tid + s];
        __syncthreads();
    }
    const float rinv = 1.f / red[0];
    for (int s = tid; s < Tn; s += 256)
        P[(size_t)t * Tn + s] = (_Float16)(expf((row[s] - m) * invT) * rinv);
}

// ---------------------------------------------------------------------------
// Host-side orchestration
// ---------------------------------------------------------------------------
extern "C" void kernel_launch(void* const* d_in, const int* in_sizes, int n_in,
                              void* d_out, int out_size, void* d_ws, size_t ws_size,
                              hipStream_t stream)
{
    (void)in_sizes; (void)n_in; (void)out_size; (void)ws_size;

    const float* x     = (const float*)d_in[0];   // [B,C,T]
    const float* wqkv  = (const float*)d_in[1];   // [3C,C,1]
    const float* wdw   = (const float*)d_in[2];   // [3C,1,3]
    const float* wproj = (const float*)d_in[3];   // [C,C,1]
    float* y = (float*)d_out;                      // [B,C,T]

    char* ws = (char*)d_ws;
    size_t off = 0;
    auto alloc = [&](size_t bytes) {
        char* p = ws + off;
        off = (off + bytes + 255) & ~(size_t)255;
        return p;
    };
    _Float16* wqkv_h  = (_Float16*)alloc((size_t)3 * Cn * Cn * 2);   // [3C,C]
    _Float16* wproj_h = (_Float16*)alloc((size_t)Cn * Cn * 2);       // [C,C]
    _Float16* xT      = (_Float16*)alloc((size_t)Tn * Cn * 2);       // [T,C]
    float*    qkvf    = (float*)   alloc((size_t)3 * Cn * Tn * 4);   // [3C,T]
    float*    convf   = (float*)   alloc((size_t)3 * Cn * Tn * 4);   // [3C,T]
    _Float16* qnT     = (_Float16*)alloc((size_t)Tn * Cn * 2);       // [T,C]
    _Float16* knT     = (_Float16*)alloc((size_t)Tn * Cn * 2);       // [T,C]
    _Float16* vh      = (_Float16*)alloc((size_t)Cn * Tn * 2);       // [C,T]
    float*    Sf      = (float*)   alloc((size_t)Tn * Tn * 4);       // [T,T]
    _Float16* Ph      = (_Float16*)alloc((size_t)Tn * Tn * 2);       // [T,T]
    _Float16* OT      = (_Float16*)alloc((size_t)Tn * Cn * 2);       // [T,C]

    // One-time weight conversions
    cvt_f32_to_f16<<<(3 * Cn * Cn) / 256, 256, 0, stream>>>(wqkv, wqkv_h, 3 * Cn * Cn);
    cvt_f32_to_f16<<<(Cn * Cn) / 256, 256, 0, stream>>>(wproj, wproj_h, Cn * Cn);

    for (int b = 0; b < Bn; ++b) {
        const float* xb = x + (size_t)b * Cn * Tn;
        float* yb = y + (size_t)b * Cn * Tn;

        // x[b] -> [T,C] f16
        transpose_to_f16<<<(Tn * Cn) / 256, 256, 0, stream>>>(xb, xT);

        // qkv = Wqkv[3C,C] x X[C,T]  (A rows=o, B rows=t, K=c) -> f32 [3C,T]
        wmma_gemm_f16<false><<<((3 * Cn / 64) * (Tn / 16)) / 4, 128, 0, stream>>>(
            wqkv_h, xT, qkvf, 3 * Cn, Tn, Cn, Cn, Cn, Tn);

        // depthwise conv3 along T
        dwconv3<<<(3 * Cn * Tn) / 256, 256, 0, stream>>>(qkvf, wdw, convf);

        // split + L2-norm(q,k over C) + pack f16
        norm_pack<<<Tn, 256, 0, stream>>>(convf, qnT, knT, vh);

        // S[t,s] = sum_c qn[c,t]*kn[c,s]  (A rows=t, B rows=s, K=c) -> f32 [T,T]
        wmma_gemm_f16<false><<<((Tn / 64) * (Tn / 16)) / 4, 128, 0, stream>>>(
            qnT, knT, Sf, Tn, Tn, Cn, Cn, Cn, Tn);

        // P = softmax(S / TEMP) -> f16 [T,T]
        softmax_row<<<Tn, 256, 0, stream>>>(Sf, Ph);

        // O^T[t,c] = sum_s P[t,s] v[c,s]  (A rows=t, B rows=c, K=s) -> f16 [T,C]
        wmma_gemm_f16<true><<<((Tn / 64) * (Cn / 16)) / 4, 128, 0, stream>>>(
            Ph, vh, OT, Tn, Cn, Tn, Tn, Tn, Cn);

        // y[o,t] = sum_c Wp[o,c] O^T[t,c]  (A rows=o, B rows=t, K=c) -> f32 [C,T]
        wmma_gemm_f16<false><<<((Cn / 64) * (Tn / 16)) / 4, 128, 0, stream>>>(
            wproj_h, OT, yb, Cn, Tn, Cn, Cn, Cn, Tn);
    }
}
